// ASAP_46772193853943
// MI455X (gfx1250) — matrix operand order repaired and checked by reading, not compiled, not run on hardware
//
#include <hip/hip_runtime.h>

#define NN 4096
#define EE 131072
#define EB (EE + NN)      // edges + self loops
#define HH 128
#define KK 3277           // ceil(0.8*N)
#define KP 3328           // K padded to multiple of 32 (and 128)

typedef unsigned short ushort_t;
typedef __attribute__((ext_vector_type(8)))  __bf16 v8bf;
typedef __attribute__((ext_vector_type(16))) __bf16 v16bf;
typedef __attribute__((ext_vector_type(8)))  float  v8f;

// ---------------- device helpers ----------------
__device__ __forceinline__ ushort_t f2bf(float f) {
    unsigned u = __float_as_uint(f);
    unsigned r = (u + 0x7FFFu + ((u >> 16) & 1u)) >> 16;   // round-nearest-even
    return (ushort_t)r;
}
__device__ __forceinline__ unsigned encf(float f) {        // order-preserving encode
    unsigned u = __float_as_uint(f);
    return (u & 0x80000000u) ? ~u : (u | 0x80000000u);
}
__device__ __forceinline__ float decf(unsigned u) {
    return (u & 0x80000000u) ? __uint_as_float(u & 0x7FFFFFFFu) : __uint_as_float(~u);
}

// ---------------- utility kernels ----------------
__global__ void k_zero_f32(float* p, int n) {
    int t = blockIdx.x * blockDim.x + threadIdx.x;
    if (t < n) p[t] = 0.0f;
}
__global__ void k_zero_i32(int* p, int n) {
    int t = blockIdx.x * blockDim.x + threadIdx.x;
    if (t < n) p[t] = 0;
}
__global__ void k_f32_bf16(ushort_t* d, const float* s, int n) {
    int t = blockIdx.x * blockDim.x + threadIdx.x;
    if (t < n) d[t] = f2bf(s[t]);
}
__global__ void k_bf16_rows(ushort_t* d, const float* s, int rows) { // [KP,HH], zero rows>=rows
    int t = blockIdx.x * blockDim.x + threadIdx.x;
    if (t >= KP * HH) return;
    int r = t >> 7;
    d[t] = (r < rows) ? f2bf(s[t]) : (ushort_t)0;
}
__global__ void k_transpose_bf16(ushort_t* dT, const float* s, int rows) { // dT[h*KP+j]=s[j*HH+h]
    int t = blockIdx.x * blockDim.x + threadIdx.x;
    if (t >= HH * KP) return;
    int h = t / KP, j = t % KP;
    dT[t] = (j < rows) ? f2bf(s[(size_t)j * HH + h]) : (ushort_t)0;
}
__global__ void k_copy_i32(int* d, const int* s, int n) {
    int t = blockIdx.x * blockDim.x + threadIdx.x;
    if (t < n) d[t] = s[t];
}

// ---------------- WMMA GEMM: out[M,128] = act(A0·B0^T (+ A1·B1^T) + bias) ----------------
// A*: bf16 row-major [M, lda*], B*: bf16 row-major [128, ldb*] (i.e. column weights).
// One wave per 16x16 tile; 8 N-tiles (N==128 always); Mtiles blocks of 256 threads.
__device__ __forceinline__ v16bf load_a_frag(const ushort_t* A, int lda, int row, int k0, int half) {
    union { v16bf v; v8bf h[2]; } u;
    const ushort_t* p = A + (size_t)row * lda + k0 + half * 8;
    u.h[0] = *(const v8bf*)p;        // K = k0+half*8 .. +7
    u.h[1] = *(const v8bf*)(p + 16); // K = k0+16+half*8 .. +7
    return u.v;
}
__device__ __forceinline__ v16bf load_b_frag(const ushort_t* B, int ldb, int col, int k0, int half) {
    union { v16bf v; v8bf h[2]; } u;
    const ushort_t* p = B + (size_t)col * ldb + k0 + half * 8;
    u.h[0] = *(const v8bf*)p;        // K = k0+half*8 .. +7
    u.h[1] = *(const v8bf*)(p + 16); // K = k0+16+half*8 .. +7
    return u.v;
}
__global__ void wmma_gemm128(const ushort_t* A0, int lda0, const ushort_t* B0, int ldb0, int K0,
                             const ushort_t* A1, int lda1, const ushort_t* B1, int ldb1, int K1,
                             const float* bias, float* out, int ldo, int relu, int Mtiles) {
    int w = (blockIdx.x * blockDim.x + threadIdx.x) >> 5;
    if (w >= Mtiles * 8) return;
    int tn = w & 7, tm = w >> 3;
    int lane = threadIdx.x & 31, half = lane >> 4, l = lane & 15;
    int m0 = tm * 16, n0 = tn * 16;
    v8f c;
    for (int r = 0; r < 8; ++r) c[r] = 0.0f;
    for (int k0 = 0; k0 < K0; k0 += 32) {
        v16bf a = load_a_frag(A0, lda0, m0 + l, k0, half);
        v16bf b = load_b_frag(B0, ldb0, n0 + l, k0, half);
        c = __builtin_amdgcn_wmma_f32_16x16x32_bf16(false, a, false, b, (short)0, c, false, false);
    }
    if (A1) {
        for (int k0 = 0; k0 < K1; k0 += 32) {
            v16bf a = load_a_frag(A1, lda1, m0 + l, k0, half);
            v16bf b = load_b_frag(B1, ldb1, n0 + l, k0, half);
            c = __builtin_amdgcn_wmma_f32_16x16x32_bf16(false, a, false, b, (short)0, c, false, false);
        }
    }
    int col = n0 + l;
    float bv = bias ? bias[col] : 0.0f;
    int rbase = m0 + half * 8;
    for (int r = 0; r < 8; ++r) {
        float v = c[r] + bv;
        if (relu) v = fmaxf(v, 0.0f);
        out[(size_t)(rbase + r) * ldo + col] = v;
    }
}

// ---------------- graph / edge kernels ----------------
__global__ void k_scatter_mean_add(const int* src, const int* dst, const float* x,
                                   float* agg, float* deg) {
    int t = blockIdx.x * blockDim.x + threadIdx.x;
    if (t >= EE * HH) return;
    int e = t >> 7, h = t & (HH - 1);
    int s = src[e], d = dst[e];
    atomicAdd(&agg[(size_t)d * HH + h], x[(size_t)s * HH + h]);
    if (h == 0) atomicAdd(&deg[d], 1.0f);
}
__global__ void k_mean_bf16(ushort_t* mb, const float* agg, const float* deg, int rows) {
    int t = blockIdx.x * blockDim.x + threadIdx.x;
    if (t >= rows * HH) return;
    int n = t >> 7;
    float dv = deg[n]; if (dv < 1.0f) dv = 1.0f;
    mb[t] = f2bf(agg[t] / dv);
}
__global__ void k_colmean(const float* x, float* out, int rows, float inv) {
    int h = threadIdx.x;              // blockDim == 128
    float s = 0.0f;
    for (int r = 0; r < rows; ++r) s += x[(size_t)r * HH + h];
    out[h] = s * inv;
}
__global__ void k_segmax(const int* src, const int* dst, const float* x2, float* xq) {
    int t = blockIdx.x * blockDim.x + threadIdx.x;
    if (t >= EB * HH) return;
    int e = t >> 7, h = t & (HH - 1);
    int j = (e < EE) ? src[e] : e - EE;
    int i = (e < EE) ? dst[e] : e - EE;
    // x2 >= 0 (post-ReLU) so int-ordered atomicMax on a zero-initialized buffer is exact
    atomicMax((int*)&xq[(size_t)i * HH + h], __float_as_int(x2[(size_t)j * HH + h]));
}
__global__ void k_node_dots(const float* xql, const float* x2, const float* Watt,
                            float* qv, float* rv) {
    int n = blockIdx.x * blockDim.x + threadIdx.x;
    if (n >= NN) return;
    float q = 0.0f, r = 0.0f;
    for (int h = 0; h < HH; ++h) {
        q += Watt[h] * xql[(size_t)n * HH + h];
        r += Watt[HH + h] * x2[(size_t)n * HH + h];
    }
    qv[n] = q; rv[n] = r;
}
__global__ void k_score1(const int* src, const int* dst, const float* qv, const float* rv,
                         const float* batt, float* sc, unsigned* menc) {
    int e = blockIdx.x * blockDim.x + threadIdx.x;
    if (e >= EB) return;
    int j = (e < EE) ? src[e] : e - EE;
    int i = (e < EE) ? dst[e] : e - EE;
    float s = qv[i] + rv[j] + batt[0];
    s = (s > 0.0f) ? s : 0.2f * s;            // leaky_relu(0.2)
    sc[e] = s;
    atomicMax(&menc[i], encf(s));
}
__global__ void k_score2(const int* dst, float* sc, const unsigned* menc, float* den) {
    int e = blockIdx.x * blockDim.x + threadIdx.x;
    if (e >= EB) return;
    int i = (e < EE) ? dst[e] : e - EE;
    float ex = expf(sc[e] - decf(menc[i]));
    sc[e] = ex;
    atomicAdd(&den[i], ex);
}
__global__ void k_norm(const int* dst, const float* sc, const float* den, float* scN) {
    int e = blockIdx.x * blockDim.x + threadIdx.x;
    if (e >= EB) return;
    int i = (e < EE) ? dst[e] : e - EE;
    scN[e] = sc[e] / den[i];
}
__global__ void k_attn(const int* src, const int* dst, const float* scN, const float* x2,
                       float* xnew) {
    int t = blockIdx.x * blockDim.x + threadIdx.x;
    if (t >= EB * HH) return;
    int e = t >> 7, h = t & (HH - 1);
    int j = (e < EE) ? src[e] : e - EE;
    int i = (e < EE) ? dst[e] : e - EE;
    atomicAdd(&xnew[(size_t)i * HH + h], scN[e] * x2[(size_t)j * HH + h]);
}
__global__ void k_fitpre(const float* xnew, const float* W1, const float* b1,
                         const float* W2, const float* W3,
                         float* av, float* bv, float* w3x) {
    int n = blockIdx.x * blockDim.x + threadIdx.x;
    if (n >= NN) return;
    float a = b1[0], b = 0.0f, c = 0.0f;
    for (int h = 0; h < HH; ++h) {
        float v = xnew[(size_t)n * HH + h];
        a += W1[h] * v; b += W2[h] * v; c += W3[h] * v;
    }
    av[n] = a; bv[n] = b; w3x[n] = c;
}
__global__ void k_fitedge(const int* src, const int* dst, const float* av, const float* bv,
                          float* aggF) {
    int e = blockIdx.x * blockDim.x + threadIdx.x;
    if (e >= EB) return;
    int j = (e < EE) ? src[e] : e - EE;
    int i = (e < EE) ? dst[e] : e - EE;
    atomicAdd(&aggF[i], av[j] - bv[i]);
}
__global__ void k_fitfinal(const float* aggF, const float* w3x, const float* b3, float* fit) {
    int n = blockIdx.x * blockDim.x + threadIdx.x;
    if (n >= NN) return;
    float z = aggF[n] + w3x[n] + b3[0];
    fit[n] = 1.0f / (1.0f + expf(-z));
}
// exact top_k (descending, ties -> lower index) via O(N^2) ranking
__global__ void k_rank(const float* fit, int* perm, float* fitk, int* invcol) {
    int n = blockIdx.x * blockDim.x + threadIdx.x;
    if (n >= NN) return;
    float f = fit[n];
    int r = 0;
    for (int m = 0; m < NN; ++m) {
        float g = fit[m];
        r += (g > f) || (g == f && m < n);
    }
    if (r < KK) { perm[r] = n; fitk[r] = f; invcol[n] = r; }
    else        { invcol[n] = -1; }
}
__global__ void k_gather_xp(const float* xnew, const int* perm, const float* fitk, float* xp) {
    int t = blockIdx.x * blockDim.x + threadIdx.x;
    if (t >= KP * HH) return;
    int p = t >> 7, h = t & (HH - 1);
    xp[t] = (p < KK) ? xnew[(size_t)perm[p] * HH + h] * fitk[p] : 0.0f;
}
// ---- CSR by source over E2 edges ----
__global__ void k_outdeg(const int* src, int* cnt) {
    int e = blockIdx.x * blockDim.x + threadIdx.x;
    if (e >= EB) return;
    int j = (e < EE) ? src[e] : e - EE;
    atomicAdd(&cnt[j], 1);
}
__global__ void k_scan(const int* cnt, int* off) {
    off[0] = 0;
    for (int n = 0; n < NN; ++n) off[n + 1] = off[n] + cnt[n];
}
__global__ void k_csrfill(const int* src, int* cursor, int* eidx) {
    int e = blockIdx.x * blockDim.x + threadIdx.x;
    if (e >= EB) return;
    int j = (e < EE) ? src[e] : e - EE;
    int p = atomicAdd(&cursor[j], 1);
    eidx[p] = e;
}
// A2[p,q] += S[j,p]*A[j,i]*S[i,q], edge-centric sparse (wave per unit of A)
__global__ void k_a2_build(const int* src, const int* dst, const int* off, const int* eidx,
                           const float* scN, const int* invcol, float* A2) {
    int w = (blockIdx.x * blockDim.x + threadIdx.x) >> 5;
    int lane = threadIdx.x & 31;
    if (w >= EB) return;
    int j = (w < EE) ? src[w] : w - EE;
    int i = (w < EE) ? dst[w] : w - EE;
    int oj = off[j], dj = off[j + 1] - oj;
    int oi = off[i], di = off[i + 1] - oi;
    int tot = dj * di;
    for (int t = lane; t < tot; t += 32) {
        int e1 = eidx[oj + t / di];
        int e2 = eidx[oi + t % di];
        int pn = (e1 < EE) ? dst[e1] : e1 - EE;
        int qn = (e2 < EE) ? dst[e2] : e2 - EE;
        int p = invcol[pn], q = invcol[qn];
        if (p >= 0 && q >= 0)
            atomicAdd(&A2[(size_t)p * KP + q], scN[e1] * scN[e2]);
    }
}
// M[i,j] = bf16(A2[j,i]) with diagonal removed; fused degree count deg[i]=#nonzero cols
__global__ void k_a2m(const float* A2, ushort_t* M, float* deg) {
    int t = blockIdx.x * blockDim.x + threadIdx.x;
    if (t >= KP * KP) return;
    int i = t / KP, j = t % KP;
    float v = 0.0f;
    if (i < KK && j < KK && i != j) v = A2[(size_t)j * KP + i];
    M[t] = f2bf(v);
    if (v != 0.0f) atomicAdd(&deg[i], 1.0f);
}

// ---------------- host side ----------------
static inline int g256(long long n) { return (int)((n + 255) / 256); }

extern "C" void kernel_launch(void* const* d_in, const int* in_sizes, int n_in,
                              void* d_out, int out_size, void* d_ws, size_t ws_size,
                              hipStream_t stream) {
    (void)in_sizes; (void)n_in; (void)out_size; (void)ws_size;
    const float* x      = (const float*)d_in[0];
    const int*   ei     = (const int*)d_in[1];
    const int*   src    = ei;
    const int*   dst    = ei + EE;
    const float* Wrel[4]  = { (const float*)d_in[2], (const float*)d_in[5],
                              (const float*)d_in[8], (const float*)d_in[11] };
    const float* brel[4]  = { (const float*)d_in[3], (const float*)d_in[6],
                              (const float*)d_in[9], (const float*)d_in[12] };
    const float* Wroot[4] = { (const float*)d_in[4], (const float*)d_in[7],
                              (const float*)d_in[10], (const float*)d_in[13] };
    const float* Wlin = (const float*)d_in[14];
    const float* blin = (const float*)d_in[15];
    const float* Watt = (const float*)d_in[16];
    const float* batt = (const float*)d_in[17];
    const float* W1   = (const float*)d_in[18];
    const float* b1   = (const float*)d_in[19];
    const float* W2   = (const float*)d_in[20];
    const float* W3   = (const float*)d_in[21];
    const float* b3   = (const float*)d_in[22];
    float* out = (float*)d_out;

    // ---- workspace carve-up ----
    char* base = (char*)d_ws; size_t off = 0;
    auto alloc = [&](size_t bytes) -> void* {
        off = (off + 255) & ~(size_t)255;
        void* p = base + off; off += bytes; return p;
    };
    ushort_t* wb[9];
    for (int i = 0; i < 9; ++i) wb[i] = (ushort_t*)alloc((size_t)HH * HH * 2);
    // wb: 0..3 = Wrel0..3, 4..7 = Wroot0..3, 8 = Wlin
    ushort_t* xb    = (ushort_t*)alloc((size_t)NN * HH * 2);
    float*    aggf  = (float*)alloc((size_t)NN * HH * 4);
    float*    degf  = (float*)alloc((size_t)NN * 4);
    ushort_t* meanb = (ushort_t*)alloc((size_t)NN * HH * 2);
    float*    x1    = (float*)alloc((size_t)NN * HH * 4);
    ushort_t* x1b   = (ushort_t*)alloc((size_t)NN * HH * 2);
    float*    x2    = (float*)alloc((size_t)NN * HH * 4);
    float*    xqf   = (float*)alloc((size_t)NN * HH * 4);
    ushort_t* xqb   = (ushort_t*)alloc((size_t)NN * HH * 2);
    float*    xql   = (float*)alloc((size_t)NN * HH * 4);
    float*    qv    = (float*)alloc((size_t)NN * 4);
    float*    rv    = (float*)alloc((size_t)NN * 4);
    float*    sc    = (float*)alloc((size_t)EB * 4);
    float*    scN   = (float*)alloc((size_t)EB * 4);
    unsigned* menc  = (unsigned*)alloc((size_t)NN * 4);
    float*    den   = (float*)alloc((size_t)NN * 4);
    float*    xnew  = (float*)alloc((size_t)NN * HH * 4);
    float*    av    = (float*)alloc((size_t)NN * 4);
    float*    bv    = (float*)alloc((size_t)NN * 4);
    float*    w3x   = (float*)alloc((size_t)NN * 4);
    float*    aggF  = (float*)alloc((size_t)NN * 4);
    float*    fitv  = (float*)alloc((size_t)NN * 4);
    int*      perm  = (int*)alloc((size_t)NN * 4);
    float*    fitk  = (float*)alloc((size_t)NN * 4);
    int*      invc  = (int*)alloc((size_t)NN * 4);
    int*      cnt   = (int*)alloc((size_t)NN * 4);
    int*      offs  = (int*)alloc((size_t)(NN + 1) * 4);
    int*      curs  = (int*)alloc((size_t)NN * 4);
    int*      eidx  = (int*)alloc((size_t)EB * 4);
    float*    A2    = (float*)alloc((size_t)KP * KP * 4);
    ushort_t* Mb    = (ushort_t*)alloc((size_t)KP * KP * 2);
    float*    xpf   = (float*)alloc((size_t)KP * HH * 4);
    ushort_t* xpb   = (ushort_t*)alloc((size_t)KP * HH * 2);
    ushort_t* xpT   = (ushort_t*)alloc((size_t)HH * KP * 2);
    float*    x3    = (float*)alloc((size_t)KP * HH * 4);
    ushort_t* x3b   = (ushort_t*)alloc((size_t)KP * HH * 2);
    ushort_t* x3T   = (ushort_t*)alloc((size_t)HH * KP * 2);
    float*    x4    = (float*)alloc((size_t)KP * HH * 4);

    const int B = 256;
    // ---- weight conversions ----
    const float* wsrc[9] = { Wrel[0], Wrel[1], Wrel[2], Wrel[3],
                             Wroot[0], Wroot[1], Wroot[2], Wroot[3], Wlin };
    for (int i = 0; i < 9; ++i)
        k_f32_bf16<<<g256(HH * HH), B, 0, stream>>>(wb[i], wsrc[i], HH * HH);
    k_f32_bf16<<<g256((long long)NN * HH), B, 0, stream>>>(xb, x, NN * HH);

    // ---- layer 1: x1 = relu(mean@Wrel0^T + brel0 + x@Wroot0^T) ----
    k_zero_f32<<<g256((long long)NN * HH), B, 0, stream>>>(aggf, NN * HH);
    k_zero_f32<<<g256(NN), B, 0, stream>>>(degf, NN);
    k_scatter_mean_add<<<g256((long long)EE * HH), B, 0, stream>>>(src, dst, x, aggf, degf);
    k_mean_bf16<<<g256((long long)NN * HH), B, 0, stream>>>(meanb, aggf, degf, NN);
    wmma_gemm128<<<NN / 16, B, 0, stream>>>(meanb, HH, wb[0], HH, HH,
                                            xb, HH, wb[4], HH, HH,
                                            brel[0], x1, HH, 1, NN / 16);
    k_colmean<<<1, HH, 0, stream>>>(x1, out + 0, NN, 1.0f / NN);
    k_f32_bf16<<<g256((long long)NN * HH), B, 0, stream>>>(x1b, x1, NN * HH);

    // ---- layer 2 ----
    k_zero_f32<<<g256((long long)NN * HH), B, 0, stream>>>(aggf, NN * HH);
    k_zero_f32<<<g256(NN), B, 0, stream>>>(degf, NN);
    k_scatter_mean_add<<<g256((long long)EE * HH), B, 0, stream>>>(src, dst, x1, aggf, degf);
    k_mean_bf16<<<g256((long long)NN * HH), B, 0, stream>>>(meanb, aggf, degf, NN);
    wmma_gemm128<<<NN / 16, B, 0, stream>>>(meanb, HH, wb[1], HH, HH,
                                            x1b, HH, wb[5], HH, HH,
                                            brel[1], x2, HH, 1, NN / 16);
    k_colmean<<<1, HH, 0, stream>>>(x2, out + HH, NN, 1.0f / NN);

    // ---- ASAP pooling ----
    k_zero_f32<<<g256((long long)NN * HH), B, 0, stream>>>(xqf, NN * HH);
    k_segmax<<<g256((long long)EB * HH), B, 0, stream>>>(src, dst, x2, xqf);
    k_f32_bf16<<<g256((long long)NN * HH), B, 0, stream>>>(xqb, xqf, NN * HH);
    wmma_gemm128<<<NN / 16, B, 0, stream>>>(xqb, HH, wb[8], HH, HH,
                                            nullptr, 0, nullptr, 0, 0,
                                            blin, xql, HH, 0, NN / 16);
    k_node_dots<<<g256(NN), B, 0, stream>>>(xql, x2, Watt, qv, rv);
    k_zero_i32<<<g256(NN), B, 0, stream>>>((int*)menc, NN);   // 0 < enc(any real) here
    k_score1<<<g256(EB), B, 0, stream>>>(src, dst, qv, rv, batt, sc, menc);
    k_zero_f32<<<g256(NN), B, 0, stream>>>(den, NN);
    k_score2<<<g256(EB), B, 0, stream>>>(dst, sc, menc, den);
    k_norm<<<g256(EB), B, 0, stream>>>(dst, sc, den, scN);
    k_zero_f32<<<g256((long long)NN * HH), B, 0, stream>>>(xnew, NN * HH);
    k_attn<<<g256((long long)EB * HH), B, 0, stream>>>(src, dst, scN, x2, xnew);
    k_fitpre<<<g256(NN), B, 0, stream>>>(xnew, W1, b1, W2, W3, av, bv, w3x);
    k_zero_f32<<<g256(NN), B, 0, stream>>>(aggF, NN);
    k_fitedge<<<g256(EB), B, 0, stream>>>(src, dst, av, bv, aggF);
    k_fitfinal<<<g256(NN), B, 0, stream>>>(aggF, w3x, b3, fitv);
    k_rank<<<g256(NN), B, 0, stream>>>(fitv, perm, fitk, invc);
    k_gather_xp<<<g256((long long)KP * HH), B, 0, stream>>>(xnew, perm, fitk, xpf);
    k_f32_bf16<<<g256((long long)KP * HH), B, 0, stream>>>(xpb, xpf, KP * HH);
    k_transpose_bf16<<<g256((long long)HH * KP), B, 0, stream>>>(xpT, xpf, KK);

    // CSR by source, then sparse A2 = S^T A S
    k_zero_i32<<<g256(NN), B, 0, stream>>>(cnt, NN);
    k_outdeg<<<g256(EB), B, 0, stream>>>(src, cnt);
    k_scan<<<1, 1, 0, stream>>>(cnt, offs);
    k_copy_i32<<<g256(NN), B, 0, stream>>>(curs, offs, NN);
    k_csrfill<<<g256(EB), B, 0, stream>>>(src, curs, eidx);
    k_zero_f32<<<g256((long long)KP * KP), B, 0, stream>>>(A2, KP * KP);
    k_a2_build<<<g256((long long)EB * 32), B, 0, stream>>>(src, dst, offs, eidx, scN, invc, A2);
    k_zero_f32<<<g256(NN), B, 0, stream>>>(degf, NN);
    k_a2m<<<g256((long long)KP * KP), B, 0, stream>>>(A2, Mb, degf);

    // ---- dense conv layer 3 ----
    wmma_gemm128<<<KP / 16, B, 0, stream>>>(Mb, KP, xpT, KP, KP,
                                            nullptr, 0, nullptr, 0, 0,
                                            nullptr, aggf, HH, 0, KP / 16);
    k_mean_bf16<<<g256((long long)KP * HH), B, 0, stream>>>(meanb, aggf, degf, KP);
    wmma_gemm128<<<KP / 16, B, 0, stream>>>(meanb, HH, wb[2], HH, HH,
                                            xpb, HH, wb[6], HH, HH,
                                            brel[2], x3, HH, 1, KP / 16);
    k_colmean<<<1, HH, 0, stream>>>(x3, out + 2 * HH, KK, 1.0f / KK);
    k_bf16_rows<<<g256((long long)KP * HH), B, 0, stream>>>(x3b, x3, KK);
    k_transpose_bf16<<<g256((long long)HH * KP), B, 0, stream>>>(x3T, x3, KK);

    // ---- dense conv layer 4 ----
    wmma_gemm128<<<KP / 16, B, 0, stream>>>(Mb, KP, x3T, KP, KP,
                                            nullptr, 0, nullptr, 0, 0,
                                            nullptr, aggf, HH, 0, KP / 16);
    k_mean_bf16<<<g256((long long)KP * HH), B, 0, stream>>>(meanb, aggf, degf, KP);
    wmma_gemm128<<<KP / 16, B, 0, stream>>>(meanb, HH, wb[3], HH, HH,
                                            x3b, HH, wb[7], HH, HH,
                                            brel[3], x4, HH, 1, KP / 16);
    k_colmean<<<1, HH, 0, stream>>>(x4, out + 3 * HH, KK, 1.0f / KK);
}
